// DynamicHyperedgeWeightLearner_37580963840550
// MI455X (gfx1250) — matrix operand order repaired and checked by reading, not compiled, not run on hardware
//
#include <hip/hip_runtime.h>
#include <hip/hip_bf16.h>
#include <math.h>

// Problem constants (from reference)
#define NNODES  8192
#define MEDGES  4096
#define DDIM    128
#define TDIM    32
#define INDIM   416      // 3*D + T_DIM = 13 * 32 (exact multiple of WMMA K)
#define H1      64
#define H2      32
#define BIGF    1e30f

// Sparsification layout: 32 n-chunks of 256 nodes, 12 slots per (edge,chunk).
// E[nnz per (edge,chunk)] = 24/32 = 0.75 ; P(Binom(256, 24/8192) >= 12) ~ 3e-11.
#define NCHUNK   32
#define CHUNK_N  (NNODES / NCHUNK)
#define CAPC     12

typedef __attribute__((ext_vector_type(8)))  __bf16 bf16x8;
typedef __attribute__((ext_vector_type(16))) __bf16 bf16x16;
typedef __attribute__((ext_vector_type(8)))  float  f32x8;

static __device__ __forceinline__ __bf16 f2bf(float f) {
  union { float f; unsigned u; } a; a.f = f;
  unsigned u = a.u + 0x7FFFu + ((a.u >> 16) & 1u);   // round-to-nearest-even
  union { unsigned short s; __bf16 b; } r; r.s = (unsigned short)(u >> 16);
  return r.b;
}

// A-matrix fragment (16-bit, 16x32): lane holds row (lane&15);
// values = K in [h*8, h*8+8) then [16+h*8, 16+h*8+8), h = lane>>4.
// -> two 16-byte loads at p and p+16 elements.
static __device__ __forceinline__ bf16x16 load_frag_a(const __bf16* p) {
  bf16x8 lo = *reinterpret_cast<const bf16x8*>(p);
  bf16x8 hi = *reinterpret_cast<const bf16x8*>(p + 16);
  return __builtin_shufflevector(lo, hi, 0,1,2,3,4,5,6,7,8,9,10,11,12,13,14,15);
}
// B-matrix fragment (16-bit, 32x16): lane holds column (lane&15);
// values = K in [h*8, h*8+8) then [16+h*8, 16+h*8+8), h = lane>>4 -> loads at p and p+16.
static __device__ __forceinline__ bf16x16 load_frag_b(const __bf16* p) {
  bf16x8 lo = *reinterpret_cast<const bf16x8*>(p);
  bf16x8 hi = *reinterpret_cast<const bf16x8*>(p + 16);
  return __builtin_shufflevector(lo, hi, 0,1,2,3,4,5,6,7,8,9,10,11,12,13,14,15);
}

// ---------------------------------------------------------------------------
// Kernel 1: convert MLP weights to bf16 (tiny, fully rewritten every call)
// ---------------------------------------------------------------------------
__global__ void prep_kernel(const float* __restrict__ W1, const float* __restrict__ W2,
                            __bf16* __restrict__ W1bf, __bf16* __restrict__ W2bf) {
  int i = blockIdx.x * blockDim.x + threadIdx.x;
  int stride = gridDim.x * blockDim.x;
  for (int j = i; j < H1 * INDIM; j += stride) W1bf[j] = f2bf(W1[j]);
  for (int j = i; j < H2 * H1;    j += stride) W2bf[j] = f2bf(W2[j]);
}

// ---------------------------------------------------------------------------
// Kernel 2: sparsify H. lane -> edge m (consecutive addresses => coalesced).
// Each thread owns one (edge, chunk) cell: deterministic, no atomics.
// This is THE memory-bound pass: 128 MiB of H @ 23.3 TB/s ~ 5.7 us.
// ---------------------------------------------------------------------------
__global__ void build_adj_kernel(const float* __restrict__ Hin,
                                 int* __restrict__ cntc, int* __restrict__ adj) {
  int m  = blockIdx.x * blockDim.x + threadIdx.x;   // 0..MEDGES-1
  int ci = blockIdx.y;                              // 0..NCHUNK-1
  int n0 = ci * CHUNK_N;
  int base = (m * NCHUNK + ci) * CAPC;
  int cnt = 0;
  for (int n = n0; n < n0 + CHUNK_N; ++n) {
    __builtin_prefetch(Hin + (size_t)(n + 16) * MEDGES + m, 0, 0);
    float v = Hin[(size_t)n * MEDGES + m];
    if (v > 0.0f) {
      if (cnt < CAPC) adj[base + cnt] = n;
      ++cnt;
    }
  }
  cntc[m * NCHUNK + ci] = cnt;
}

// ---------------------------------------------------------------------------
// Kernel 3: per-edge stats. One wave per edge, 4 dims per lane (float4).
// Single pass over ~24 member rows of x (L2-resident) computes sum, sumsq,
// max, min -> mu | sigma | delta | t_embed written as one bf16 row of h.
// ---------------------------------------------------------------------------
__global__ void edge_stats_kernel(const float* __restrict__ x,
                                  const int* __restrict__ cntc,
                                  const int* __restrict__ adj,
                                  const float* __restrict__ t,
                                  const float* __restrict__ Wt,
                                  const float* __restrict__ bt,
                                  __bf16* __restrict__ Hbf) {
  int edge = blockIdx.x * (blockDim.x >> 5) + (threadIdx.x >> 5);
  int lane = threadIdx.x & 31;
  if (edge >= MEDGES) return;

  const int* cbase = cntc + edge * NCHUNK;
  int tot = 0;
  for (int ci = 0; ci < NCHUNK; ++ci) tot += cbase[ci];

  float s0 = 0.f, s1 = 0.f, s2 = 0.f, s3 = 0.f;
  float q0 = 0.f, q1 = 0.f, q2 = 0.f, q3 = 0.f;
  float mx0 = -BIGF, mx1 = -BIGF, mx2 = -BIGF, mx3 = -BIGF;
  float mn0 =  BIGF, mn1 =  BIGF, mn2 =  BIGF, mn3 =  BIGF;

  for (int ci = 0; ci < NCHUNK; ++ci) {
    int kc = cbase[ci]; kc = kc < CAPC ? kc : CAPC;
    const int* lst = adj + (edge * NCHUNK + ci) * CAPC;
    for (int j = 0; j < kc; ++j) {
      int n = lst[j];  // uniform across the wave
      const float4 v = *reinterpret_cast<const float4*>(x + (size_t)n * DDIM + lane * 4);
      s0 += v.x; s1 += v.y; s2 += v.z; s3 += v.w;
      q0 += v.x * v.x; q1 += v.y * v.y; q2 += v.z * v.z; q3 += v.w * v.w;
      mx0 = fmaxf(mx0, v.x); mx1 = fmaxf(mx1, v.y); mx2 = fmaxf(mx2, v.z); mx3 = fmaxf(mx3, v.w);
      mn0 = fminf(mn0, v.x); mn1 = fminf(mn1, v.y); mn2 = fminf(mn2, v.z); mn3 = fminf(mn3, v.w);
    }
  }

  float deg = (float)(tot > 0 ? tot : 1);
  float inv = 1.0f / deg;
  float mu0 = s0 * inv, mu1 = s1 * inv, mu2 = s2 * inv, mu3 = s3 * inv;
  float sg0 = sqrtf(fmaxf(q0 * inv - mu0 * mu0, 1e-8f));
  float sg1 = sqrtf(fmaxf(q1 * inv - mu1 * mu1, 1e-8f));
  float sg2 = sqrtf(fmaxf(q2 * inv - mu2 * mu2, 1e-8f));
  float sg3 = sqrtf(fmaxf(q3 * inv - mu3 * mu3, 1e-8f));
  float dl0 = tot > 0 ? (mx0 - mn0) : 0.f;
  float dl1 = tot > 0 ? (mx1 - mn1) : 0.f;
  float dl2 = tot > 0 ? (mx2 - mn2) : 0.f;
  float dl3 = tot > 0 ? (mx3 - mn3) : 0.f;

  __bf16* row = Hbf + (size_t)edge * INDIM;
  int o = lane * 4;
  row[o + 0] = f2bf(mu0); row[o + 1] = f2bf(mu1); row[o + 2] = f2bf(mu2); row[o + 3] = f2bf(mu3);
  row[DDIM + o + 0] = f2bf(sg0); row[DDIM + o + 1] = f2bf(sg1);
  row[DDIM + o + 2] = f2bf(sg2); row[DDIM + o + 3] = f2bf(sg3);
  row[2 * DDIM + o + 0] = f2bf(dl0); row[2 * DDIM + o + 1] = f2bf(dl1);
  row[2 * DDIM + o + 2] = f2bf(dl2); row[2 * DDIM + o + 3] = f2bf(dl3);
  // t_embed: one of TDIM=32 features per lane
  float te = fmaxf(t[0] * Wt[lane] + bt[lane], 0.0f);
  row[3 * DDIM + lane] = f2bf(te);
}

// ---------------------------------------------------------------------------
// Kernel 4: fused 3-layer MLP via bf16 WMMA. One wave per 16-edge tile.
// L1: (16x416)@(416x64)  = 13 k-steps x 4 n-tiles of v_wmma_f32_16x16x32_bf16
// L2: (16x64)@(64x32)    =  2 k-steps x 2 n-tiles (A staged through LDS)
// L3: (16x32)@(32x1) + sigmoid in VALU.
// ---------------------------------------------------------------------------
__global__ void __launch_bounds__(32) mlp_kernel(const __bf16* __restrict__ Hbf,
                                                 const __bf16* __restrict__ W1bf,
                                                 const __bf16* __restrict__ W2bf,
                                                 const float* __restrict__ b1,
                                                 const float* __restrict__ b2,
                                                 const float* __restrict__ W3,
                                                 const float* __restrict__ b3,
                                                 float* __restrict__ out) {
  const int m0 = blockIdx.x * 16;
  const int l  = threadIdx.x;
  const int lo = l & 15;
  const int hi = l >> 4;

  __shared__ __align__(16) __bf16 A1[16 * H1];   // layer-1 activations (bf16)
  __shared__ float A2[16 * H2];                  // layer-2 activations (f32)

  // ---- Layer 1 ----
  f32x8 acc[4] = {};
#pragma unroll
  for (int kt = 0; kt < 13; ++kt) {
    bf16x16 a = load_frag_a(Hbf + (size_t)(m0 + lo) * INDIM + kt * 32 + hi * 8);
#pragma unroll
    for (int nt = 0; nt < 4; ++nt) {
      bf16x16 b = load_frag_b(W1bf + (size_t)(nt * 16 + lo) * INDIM + kt * 32 + hi * 8);
      acc[nt] = __builtin_amdgcn_wmma_f32_16x16x32_bf16(
          false, a, false, b, (short)0, acc[nt], false, false);
    }
  }
  // C/D layout: VGPR r, lane l -> (row = r + hi*8, col = lo) within the tile
#pragma unroll
  for (int nt = 0; nt < 4; ++nt) {
#pragma unroll
    for (int r = 0; r < 8; ++r) {
      int rowm = r + hi * 8, col = nt * 16 + lo;
      A1[rowm * H1 + col] = f2bf(fmaxf(acc[nt][r] + b1[col], 0.0f));
    }
  }
  __syncthreads();

  // ---- Layer 2 ----
  f32x8 acc2[2] = {};
#pragma unroll
  for (int kt = 0; kt < 2; ++kt) {
    bf16x16 a = load_frag_a(&A1[lo * H1 + kt * 32 + hi * 8]);
#pragma unroll
    for (int nt = 0; nt < 2; ++nt) {
      bf16x16 b = load_frag_b(W2bf + (nt * 16 + lo) * H1 + kt * 32 + hi * 8);
      acc2[nt] = __builtin_amdgcn_wmma_f32_16x16x32_bf16(
          false, a, false, b, (short)0, acc2[nt], false, false);
    }
  }
#pragma unroll
  for (int nt = 0; nt < 2; ++nt) {
#pragma unroll
    for (int r = 0; r < 8; ++r) {
      int rowm = r + hi * 8, col = nt * 16 + lo;
      A2[rowm * H2 + col] = fmaxf(acc2[nt][r] + b2[col], 0.0f);
    }
  }
  __syncthreads();

  // ---- Layer 3 + sigmoid ----
  if (l < 16) {
    float s = b3[0];
#pragma unroll
    for (int j = 0; j < H2; ++j) s += A2[l * H2 + j] * W3[j];
    out[m0 + l] = 1.0f / (1.0f + expf(-s));
  }
}

// ---------------------------------------------------------------------------
extern "C" void kernel_launch(void* const* d_in, const int* in_sizes, int n_in,
                              void* d_out, int out_size, void* d_ws, size_t ws_size,
                              hipStream_t stream) {
  (void)in_sizes; (void)n_in; (void)out_size; (void)ws_size;
  const float* x   = (const float*)d_in[0];   // (N, D)
  const float* Hin = (const float*)d_in[1];   // (N, M)
  const float* t   = (const float*)d_in[2];   // (1,)
  const float* Wt  = (const float*)d_in[3];   // (32, 1)
  const float* bt  = (const float*)d_in[4];   // (32,)
  const float* W1  = (const float*)d_in[5];   // (64, 416)
  const float* b1  = (const float*)d_in[6];   // (64,)
  const float* W2  = (const float*)d_in[7];   // (32, 64)
  const float* b2  = (const float*)d_in[8];   // (32,)
  const float* W3  = (const float*)d_in[9];   // (1, 32)
  const float* b3  = (const float*)d_in[10];  // (1,)
  float* out = (float*)d_out;

  // Workspace layout (all regions fully rewritten every call; ~9.9 MB)
  char* ws = (char*)d_ws;
  int*    cntc = (int*)ws;                                   //   512 KiB
  int*    adj  = (int*)(ws + 524288);                        //  6144 KiB
  __bf16* Hbf  = (__bf16*)(ws + 524288 + 6291456);           //  3328 KiB
  __bf16* W1bf = (__bf16*)(ws + 524288 + 6291456 + 3407872); //    52 KiB
  __bf16* W2bf = (__bf16*)(ws + 524288 + 6291456 + 3407872 + 53248); // 4 KiB

  prep_kernel<<<32, 256, 0, stream>>>(W1, W2, W1bf, W2bf);
  build_adj_kernel<<<dim3(MEDGES / 256, NCHUNK), 256, 0, stream>>>(Hin, cntc, adj);
  edge_stats_kernel<<<MEDGES / 4, 128, 0, stream>>>(x, cntc, adj, t, Wt, bt, Hbf);
  mlp_kernel<<<MEDGES / 16, 32, 0, stream>>>(Hbf, W1bf, W2bf, b1, b2, W3, b3, out);
}
